// MMFP4MoE_30915174596903
// MI455X (gfx1250) — compile-verified, matrix operand order checked by reading, not run on hardware
//
#include <hip/hip_runtime.h>
#include <cstdint>

// ---- problem constants (from reference) ----
#define HH   2048   // hidden
#define II   1536   // intermediate
#define NE   64     // experts
#define TKK  4      // top-k
#define CAP  128    // capacity per expert
#define NT   1024   // tokens
#define GRP  128    // quant group size
#define LDB  40     // padded LDS row stride (halfs): 80B -> conflict-free b128

typedef _Float16 v16h __attribute__((ext_vector_type(16)));
typedef _Float16 v8h  __attribute__((ext_vector_type(8)));
typedef _Float16 h2v  __attribute__((ext_vector_type(2)));
typedef float    v8f  __attribute__((ext_vector_type(8)));

// LDS tile sizes (halfs / bytes)
#define ASZH (128 * LDB)      // 5120 halfs = 10240 B
#define BSZH (64 * LDB)       // 2560 halfs = 5120 B
#define ASZB (ASZH * 2)
#define BSZB (BSZH * 2)

// ---------------- CDNA5 async-to-LDS helpers ----------------
// GLOBAL_LOAD_ASYNC_TO_LDS_B128: per-lane 16B global->LDS DMA, tracked by
// ASYNCcnt. ldsOff = per-lane LDS byte address (dynamic LDS base == 0 here,
// no static __shared__ in these kernels).
__device__ __forceinline__ void async_ld_b128(uint32_t ldsOff, const _Float16* g) {
  asm volatile("global_load_async_to_lds_b128 %0, %1, off"
               :: "v"(ldsOff), "v"(g) : "memory");
}
__device__ __forceinline__ void wait_async() {
  asm volatile("s_wait_asynccnt 0" ::: "memory");
}

// ---------------- FP4 -> f16 dequant helpers ----------------
// FP4 E2M1 magnitudes as f16 high bytes (low bytes all 0x00):
//   idx: 0     1     2     3     4     5     6     7
//   f16: 0000  3800  3C00  3E00  4000  4200  4400  4600
// One v_perm_b32 covers 4 nibbles' high bytes; sign = nibble bit3 -> f16 bit15.
__device__ __forceinline__ uint32_t pk_mul(uint32_t a, uint32_t s) {
  h2v x = __builtin_bit_cast(h2v, a);
  h2v y = __builtin_bit_cast(h2v, s);
  h2v r = x * y;
  return __builtin_bit_cast(uint32_t, r);
}

__device__ __forceinline__ uint32_t pack_scale(float s) {
  _Float16 h = (_Float16)s;
  uint16_t b = __builtin_bit_cast(uint16_t, h);
  return ((uint32_t)b << 16) | (uint32_t)b;
}

// w: 8 FP4 nibbles (nibble j = element K=j). sp: packed half2 scale.
__device__ __forceinline__ void dequant8_scale(uint32_t w, uint32_t sp, uint32_t o[4]) {
  uint32_t sel_lo = w & 0x07070707u;          // mags of nibbles 0,2,4,6
  uint32_t sel_hi = (w >> 4) & 0x07070707u;   // mags of nibbles 1,3,5,7
  uint32_t hb_lo = __builtin_amdgcn_perm(0x46444240u, 0x3E3C3800u, sel_lo)
                 | ((w << 4) & 0x80808080u);
  uint32_t hb_hi = __builtin_amdgcn_perm(0x46444240u, 0x3E3C3800u, sel_hi)
                 | (w & 0x80808080u);
  uint32_t d0 = ((hb_lo & 0x000000FFu) << 8)  | ((hb_hi & 0x000000FFu) << 24);
  uint32_t d1 = ( hb_lo & 0x0000FF00u)        | ((hb_hi & 0x0000FF00u) << 16);
  uint32_t d2 = ((hb_lo & 0x00FF0000u) >> 8)  | ((hb_hi & 0x00FF0000u) << 8);
  uint32_t d3 = ((hb_lo & 0xFF000000u) >> 16) | ( hb_hi & 0xFF000000u);
  o[0] = pk_mul(d0, sp); o[1] = pk_mul(d1, sp);
  o[2] = pk_mul(d2, sp); o[3] = pk_mul(d3, sp);
}

__device__ __forceinline__ void store_b128(_Float16* dst, const uint32_t o[4]) {
  uint4 v; v.x = o[0]; v.y = o[1]; v.z = o[2]; v.w = o[3];
  *(uint4*)dst = v;
}

// ---------------- WMMA fragment loader ----------------
// 16-bit A/B 16x32 layout: lane L holds row (L&15); VGPR0-3 = K[half*8 .. +7],
// VGPR4-7 = K[16+half*8 .. +7], half = L>>4. Tile rows padded to LDB halfs.
__device__ __forceinline__ v16h ld_frag(const _Float16* base, int rowBase, int lane) {
  int r = rowBase + (lane & 15);
  const _Float16* p = base + r * LDB + ((lane >> 4) << 3);
  v8h lo = *(const v8h*)p;
  v8h hi = *(const v8h*)(p + 16);
  return __builtin_shufflevector(lo, hi, 0,1,2,3,4,5,6,7,8,9,10,11,12,13,14,15);
}

#define WMMA_F16(A, B, C) \
  __builtin_amdgcn_wmma_f32_16x16x32_f16(false, (A), false, (B), (short)0, (C), false, false)

// ---------------- utility kernels ----------------
__global__ void fill_u32(uint32_t* p, uint32_t v, size_t n) {
  size_t i = (size_t)blockIdx.x * blockDim.x + threadIdx.x;
  size_t stride = (size_t)gridDim.x * blockDim.x;
  for (; i < n; i += stride) p[i] = v;
}

__global__ void xcvt_kernel(const float* __restrict__ x, _Float16* __restrict__ xsh) {
  size_t i = (size_t)blockIdx.x * blockDim.x + threadIdx.x;
  size_t stride = (size_t)gridDim.x * blockDim.x;
  size_t n = (size_t)NT * HH;
  for (; i < n; i += stride) xsh[i] = (_Float16)x[i];
}

// ---------------- router: logits, top-4, slot assignment ----------------
__global__ void router_kernel(const float* __restrict__ x, const float* __restrict__ gw,
                              int* __restrict__ counts, int* __restrict__ slot_map,
                              float* __restrict__ wt) {
  __shared__ float xs[HH];
  __shared__ float lg[NE];
  int t = blockIdx.x;
  for (int i = threadIdx.x; i < HH; i += 64) xs[i] = x[(size_t)t * HH + i];
  __syncthreads();
  {
    int e = threadIdx.x;  // blockDim.x == NE == 64
    const float* w = gw + (size_t)e * HH;
    float acc = 0.f;
    for (int i = 0; i < HH; ++i) acc += xs[i] * w[i];
    lg[e] = acc;
  }
  __syncthreads();
  if (threadIdx.x == 0) {
    int   idx[TKK];
    float val[TKK];
    for (int k = 0; k < TKK; ++k) {
      float best = -1e30f; int bi = 0;
      for (int j = 0; j < NE; ++j) {
        bool used = false;
        for (int p = 0; p < k; ++p) used |= (idx[p] == j);
        if (!used && lg[j] > best) { best = lg[j]; bi = j; }
      }
      idx[k] = bi; val[k] = best;
    }
    // softmax denominator cancels under top-k renormalization
    float mx = val[0], s = 0.f, wv[TKK];
    for (int k = 0; k < TKK; ++k) { wv[k] = __expf(val[k] - mx); s += wv[k]; }
    float inv = 1.f / s;
    for (int k = 0; k < TKK; ++k) {
      int a = t * TKK + k;
      wt[a] = wv[k] * inv;
      int pos = atomicAdd(&counts[idx[k]], 1);
      slot_map[a] = (pos < CAP) ? (idx[k] * CAP + pos) : -1;
    }
  }
}

// ---------------- scatter tokens into expert buffers ----------------
__global__ void scatter_kernel(const float* __restrict__ x, const int* __restrict__ slot_map,
                               const float* __restrict__ wt, _Float16* __restrict__ xbuf,
                               int* __restrict__ slot_tok, float* __restrict__ slot_w) {
  int a = blockIdx.x;
  int slot = slot_map[a];
  if (slot < 0) return;
  int t = a >> 2;
  for (int i = threadIdx.x; i < HH; i += blockDim.x)
    xbuf[(size_t)slot * HH + i] = (_Float16)x[(size_t)t * HH + i];
  if (threadIdx.x == 0) { slot_tok[slot] = t; slot_w[slot] = wt[a]; }
}

// ---------------- fused gate+up GEMM with SwiGLU epilogue ----------------
// Block: 256 thr = 8 waves (4 M-tiles x 2 N-tiles); tile 128x64, K-step 32.
// 2-stage pipeline: async-to-LDS A tiles + register-staged dequant B tiles,
// ping-pong LDS buffers, one s_wait_asynccnt + barrier per K-step.
// Dynamic LDS layout (bytes): [A0 10240][A1 10240][Bg0 5120][Bg1 5120][Bu0][Bu1]
__global__ __launch_bounds__(256) void gateup_kernel(
    const _Float16* __restrict__ A, size_t aStride,
    const int* __restrict__ Bg, const int* __restrict__ Bu, size_t bStride,
    const float* __restrict__ Sg, const float* __restrict__ Su, size_t sStride,
    _Float16* __restrict__ Hout, size_t hStride) {
  extern __shared__ __align__(16) char smem[];
  _Float16* SM = (_Float16*)smem;

  const int tid = threadIdx.x, lane = tid & 31;
  const int wv = tid >> 5, wm = wv & 3, wn = wv >> 2;
  const int colBase = blockIdx.x * 64;
  const int ey = blockIdx.y;
  const _Float16* Ae = A + (size_t)ey * aStride;
  const int* bg = Bg + (size_t)ey * bStride;
  const int* bu = Bu + (size_t)ey * bStride;
  const float* sg = Sg + (size_t)ey * sStride;
  const float* su = Su + (size_t)ey * sStride;
  _Float16* ho = Hout + (size_t)ey * hStride;

  v8f accG[2][2] = {{{}, {}}, {{}, {}}};
  v8f accU[2][2] = {{{}, {}}, {{}, {}}};
  const int arow = tid >> 2, aoff = (tid & 3) * 8;
  const int brow = tid >> 2, bdw = tid & 3;
  const int col = colBase + brow;
  const uint32_t aLds0 = (uint32_t)((arow * LDB + aoff) * 2);
  const uint32_t aLds1 = (uint32_t)(((arow + 64) * LDB + aoff) * 2);
  const int bIdx = brow * LDB + bdw * 8;
  constexpr int NKT = HH / 32;

  // ---- prologue: tile 0 into buffer 0 ----
  async_ld_b128(aLds0, Ae + (size_t)arow * HH + aoff);
  async_ld_b128(aLds1, Ae + (size_t)(arow + 64) * HH + aoff);
  {
    uint32_t wg = (uint32_t)bg[(size_t)col * (HH / 8) + bdw];
    uint32_t wu = (uint32_t)bu[(size_t)col * (HH / 8) + bdw];
    uint32_t o4[4];
    dequant8_scale(wg, pack_scale(sg[col]), o4);
    store_b128(SM + 2 * ASZH + bIdx, o4);
    dequant8_scale(wu, pack_scale(su[col]), o4);
    store_b128(SM + 2 * ASZH + 2 * BSZH + bIdx, o4);
  }
  wait_async();
  __syncthreads();

  int p = 0;
  for (int kt = 0; kt < NKT; ++kt) {
    const int q = p ^ 1;
    uint32_t wgN = 0, wuN = 0, spgN = 0, spuN = 0;
    const bool havN = (kt + 1 < NKT);
    if (havN) {
      const int kn = kt + 1, g = kn >> 2;  // GS=128 -> group per 4 K-steps
      async_ld_b128((uint32_t)(q * ASZB) + aLds0, Ae + (size_t)arow * HH + kn * 32 + aoff);
      async_ld_b128((uint32_t)(q * ASZB) + aLds1, Ae + (size_t)(arow + 64) * HH + kn * 32 + aoff);
      wgN = (uint32_t)bg[(size_t)col * (HH / 8) + kn * 4 + bdw];
      wuN = (uint32_t)bu[(size_t)col * (HH / 8) + kn * 4 + bdw];
      spgN = pack_scale(sg[g * II + col]);
      spuN = pack_scale(su[g * II + col]);
      if (kt + 2 < NKT) {
        __builtin_prefetch(&bg[(size_t)col * (HH / 8) + (kt + 2) * 4 + bdw], 0, 1);
        __builtin_prefetch(&bu[(size_t)col * (HH / 8) + (kt + 2) * 4 + bdw], 0, 1);
      }
    }
    // ---- compute K-step kt from buffer p ----
    const _Float16* Ap  = SM + p * ASZH;
    const _Float16* Bgp = SM + 2 * ASZH + p * BSZH;
    const _Float16* Bup = SM + 2 * ASZH + 2 * BSZH + p * BSZH;
    v16h a0 = ld_frag(Ap,  wm * 32 + 0,  lane);
    v16h a1 = ld_frag(Ap,  wm * 32 + 16, lane);
    v16h g0 = ld_frag(Bgp, wn * 32 + 0,  lane);
    v16h g1 = ld_frag(Bgp, wn * 32 + 16, lane);
    v16h u0 = ld_frag(Bup, wn * 32 + 0,  lane);
    v16h u1 = ld_frag(Bup, wn * 32 + 16, lane);
    accG[0][0] = WMMA_F16(a0, g0, accG[0][0]);
    accG[0][1] = WMMA_F16(a0, g1, accG[0][1]);
    accG[1][0] = WMMA_F16(a1, g0, accG[1][0]);
    accG[1][1] = WMMA_F16(a1, g1, accG[1][1]);
    accU[0][0] = WMMA_F16(a0, u0, accU[0][0]);
    accU[0][1] = WMMA_F16(a0, u1, accU[0][1]);
    accU[1][0] = WMMA_F16(a1, u0, accU[1][0]);
    accU[1][1] = WMMA_F16(a1, u1, accU[1][1]);
    // ---- stage next B tiles into buffer q ----
    if (havN) {
      uint32_t o4[4];
      dequant8_scale(wgN, spgN, o4);
      store_b128(SM + 2 * ASZH + q * BSZH + bIdx, o4);
      dequant8_scale(wuN, spuN, o4);
      store_b128(SM + 2 * ASZH + 2 * BSZH + q * BSZH + bIdx, o4);
    }
    wait_async();
    __syncthreads();
    p = q;
  }
  // SwiGLU epilogue: h = silu(g) * u
  for (int i = 0; i < 2; ++i)
    for (int j = 0; j < 2; ++j) {
      v8f g = accG[i][j], u = accU[i][j];
      int n = colBase + wn * 32 + j * 16 + (lane & 15);
      int mBase = wm * 32 + i * 16 + ((lane >> 4) << 3);
      for (int r = 0; r < 8; ++r) {
        float gg = g[r], uu = u[r];
        float hv = gg * uu / (1.0f + __expf(-gg));
        ho[(size_t)(mBase + r) * II + n] = (_Float16)hv;
      }
    }
}

// ---------------- down GEMM + combine ----------------
// Dynamic LDS layout (bytes): [A0 10240][A1 10240][B0 5120][B1 5120]
template <bool EXPERT>
__global__ __launch_bounds__(256) void down_kernel(
    const _Float16* __restrict__ A, size_t aStride,
    const int* __restrict__ Bd, size_t bStride,
    const float* __restrict__ Sd, size_t sStride,
    float* __restrict__ out,
    const int* __restrict__ slot_tok, const float* __restrict__ slot_w) {
  extern __shared__ __align__(16) char smem[];
  _Float16* SM = (_Float16*)smem;

  const int tid = threadIdx.x, lane = tid & 31;
  const int wv = tid >> 5, wm = wv & 3, wn = wv >> 2;
  const int colBase = blockIdx.x * 64;  // over HH
  const int ey = blockIdx.y;
  const _Float16* Ae = A + (size_t)ey * aStride;
  const int* bd = Bd + (size_t)ey * bStride;
  const float* sd = Sd + (size_t)ey * sStride;

  v8f acc[2][2] = {{{}, {}}, {{}, {}}};
  const int arow = tid >> 2, aoff = (tid & 3) * 8;
  const int brow = tid >> 2, bdw = tid & 3;
  const int col = colBase + brow;
  const uint32_t aLds0 = (uint32_t)((arow * LDB + aoff) * 2);
  const uint32_t aLds1 = (uint32_t)(((arow + 64) * LDB + aoff) * 2);
  const int bIdx = brow * LDB + bdw * 8;
  constexpr int NKT = II / 32;

  // ---- prologue ----
  async_ld_b128(aLds0, Ae + (size_t)arow * II + aoff);
  async_ld_b128(aLds1, Ae + (size_t)(arow + 64) * II + aoff);
  {
    uint32_t w = (uint32_t)bd[(size_t)col * (II / 8) + bdw];
    uint32_t o4[4];
    dequant8_scale(w, pack_scale(sd[col]), o4);
    store_b128(SM + 2 * ASZH + bIdx, o4);
  }
  wait_async();
  __syncthreads();

  int p = 0;
  for (int kt = 0; kt < NKT; ++kt) {
    const int q = p ^ 1;
    uint32_t wN = 0, spN = 0;
    const bool havN = (kt + 1 < NKT);
    if (havN) {
      const int kn = kt + 1, g = kn >> 2;
      async_ld_b128((uint32_t)(q * ASZB) + aLds0, Ae + (size_t)arow * II + kn * 32 + aoff);
      async_ld_b128((uint32_t)(q * ASZB) + aLds1, Ae + (size_t)(arow + 64) * II + kn * 32 + aoff);
      wN  = (uint32_t)bd[(size_t)col * (II / 8) + kn * 4 + bdw];
      spN = pack_scale(sd[g * HH + col]);
      if (kt + 2 < NKT)
        __builtin_prefetch(&bd[(size_t)col * (II / 8) + (kt + 2) * 4 + bdw], 0, 1);
    }
    const _Float16* Ap = SM + p * ASZH;
    const _Float16* Bp = SM + 2 * ASZH + p * BSZH;
    v16h a0 = ld_frag(Ap, wm * 32 + 0,  lane);
    v16h a1 = ld_frag(Ap, wm * 32 + 16, lane);
    v16h b0 = ld_frag(Bp, wn * 32 + 0,  lane);
    v16h b1 = ld_frag(Bp, wn * 32 + 16, lane);
    acc[0][0] = WMMA_F16(a0, b0, acc[0][0]);
    acc[0][1] = WMMA_F16(a0, b1, acc[0][1]);
    acc[1][0] = WMMA_F16(a1, b0, acc[1][0]);
    acc[1][1] = WMMA_F16(a1, b1, acc[1][1]);
    if (havN) {
      uint32_t o4[4];
      dequant8_scale(wN, spN, o4);
      store_b128(SM + 2 * ASZH + q * BSZH + bIdx, o4);
    }
    wait_async();
    __syncthreads();
    p = q;
  }
  for (int i = 0; i < 2; ++i)
    for (int j = 0; j < 2; ++j) {
      v8f y = acc[i][j];
      int n = colBase + wn * 32 + j * 16 + (lane & 15);
      int mBase = wm * 32 + i * 16 + ((lane >> 4) << 3);
      for (int r = 0; r < 8; ++r) {
        int m = mBase + r;
        if (EXPERT) {
          int tok = slot_tok[ey * CAP + m];
          if (tok >= 0) {
            float wgt = slot_w[ey * CAP + m];
            __hip_atomic_fetch_add(&out[(size_t)tok * HH + n], wgt * y[r],
                                   __ATOMIC_RELAXED, __HIP_MEMORY_SCOPE_AGENT);
          }
        } else {
          out[(size_t)(ey * 128 + m) * HH + n] = y[r];
        }
      }
    }
}

// ---------------- host launch ----------------
extern "C" void kernel_launch(void* const* d_in, const int* in_sizes, int n_in,
                              void* d_out, int out_size, void* d_ws, size_t ws_size,
                              hipStream_t stream) {
  const float* x              = (const float*)d_in[0];
  const float* gate_w         = (const float*)d_in[1];
  const float* gate_scales    = (const float*)d_in[2];
  const float* up_scales      = (const float*)d_in[3];
  const float* down_scales    = (const float*)d_in[4];
  const float* sh_gate_scales = (const float*)d_in[5];
  const float* sh_up_scales   = (const float*)d_in[6];
  const float* sh_down_scales = (const float*)d_in[7];
  const int*   gate_packed    = (const int*)d_in[8];
  const int*   up_packed      = (const int*)d_in[9];
  const int*   down_packed    = (const int*)d_in[10];
  const int*   sh_gate_packed = (const int*)d_in[11];
  const int*   sh_up_packed   = (const int*)d_in[12];
  const int*   sh_down_packed = (const int*)d_in[13];
  float* out = (float*)d_out;

  // workspace carve (256B aligned)
  char* ws = (char*)d_ws;
  size_t off = 0;
  auto carve = [&](size_t bytes) {
    void* p = ws + off;
    off = (off + bytes + 255) & ~(size_t)255;
    return p;
  };
  int*      counts   = (int*)carve(NE * sizeof(int));
  int*      slot_map = (int*)carve((size_t)NT * TKK * sizeof(int));
  float*    wt       = (float*)carve((size_t)NT * TKK * sizeof(float));
  int*      slot_tok = (int*)carve((size_t)NE * CAP * sizeof(int));
  float*    slot_w   = (float*)carve((size_t)NE * CAP * sizeof(float));
  _Float16* xbuf     = (_Float16*)carve((size_t)NE * CAP * HH * 2);
  _Float16* xsh      = (_Float16*)carve((size_t)NT * HH * 2);
  _Float16* Hbuf     = (_Float16*)carve((size_t)NE * CAP * II * 2);
  _Float16* Hsh      = (_Float16*)carve((size_t)NT * II * 2);
  (void)ws_size; (void)in_sizes; (void)n_in; (void)out_size;

  const size_t ldsGU = 2 * (size_t)ASZB + 4 * (size_t)BSZB;  // 40960 B
  const size_t ldsDN = 2 * (size_t)ASZB + 2 * (size_t)BSZB;  // 30720 B

  // 1) init
  fill_u32<<<64, 256, 0, stream>>>((uint32_t*)counts, 0u, NE);
  fill_u32<<<64, 256, 0, stream>>>((uint32_t*)slot_tok, 0xFFFFFFFFu, (size_t)NE * CAP);
  fill_u32<<<2048, 256, 0, stream>>>((uint32_t*)xbuf, 0u, (size_t)NE * CAP * HH / 2);

  // 2) router
  router_kernel<<<NT, 64, 0, stream>>>(x, gate_w, counts, slot_map, wt);

  // 3) dispatch + shared-path activation convert
  xcvt_kernel<<<2048, 256, 0, stream>>>(x, xsh);
  scatter_kernel<<<NT * TKK, 256, 0, stream>>>(x, slot_map, wt, xbuf, slot_tok, slot_w);

  // 4) shared expert (writes out directly, must precede expert atomics)
  gateup_kernel<<<dim3(II / 64, NT / 128), 256, ldsGU, stream>>>(
      xsh, (size_t)128 * HH,
      sh_gate_packed, sh_up_packed, 0,
      sh_gate_scales, sh_up_scales, 0,
      Hsh, (size_t)128 * II);
  down_kernel<false><<<dim3(HH / 64, NT / 128), 256, ldsDN, stream>>>(
      Hsh, (size_t)128 * II,
      sh_down_packed, 0,
      sh_down_scales, 0,
      out, nullptr, nullptr);

  // 5) routed experts (atomic-add weighted contributions)
  gateup_kernel<<<dim3(II / 64, NE), 256, ldsGU, stream>>>(
      xbuf, (size_t)CAP * HH,
      gate_packed, up_packed, (size_t)II * (HH / 8),
      gate_scales, up_scales, (size_t)(HH / GRP) * II,
      Hbuf, (size_t)CAP * II);
  down_kernel<true><<<dim3(HH / 64, NE), 256, ldsDN, stream>>>(
      Hbuf, (size_t)CAP * II,
      down_packed, (size_t)HH * (II / 8),
      down_scales, (size_t)(II / GRP) * HH,
      out, slot_tok, slot_w);
}
